// EfficientVitBlockND_88218628260014
// MI455X (gfx1250) — matrix operand order repaired and checked by reading, not compiled, not run on hardware
//
#include <hip/hip_runtime.h>

// ---------- types ----------
typedef __attribute__((ext_vector_type(16))) __bf16        v16bf;
typedef __attribute__((ext_vector_type(8)))  float         v8f;
typedef __attribute__((ext_vector_type(4)))  unsigned int  u32x4;
typedef __attribute__((ext_vector_type(8)))  unsigned short u16x8;
typedef unsigned short us16;
typedef unsigned int   u32;

union Frag { u32x4 q[2]; v16bf v; };

// ---------- scalar helpers ----------
__device__ __forceinline__ us16 f2bf(float f) {
  u32 u = __float_as_uint(f);
  u32 r = u + 0x7FFFu + ((u >> 16) & 1u);
  return (us16)(r >> 16);
}
__device__ __forceinline__ float bf2f(us16 h) {
  return __uint_as_float(((u32)h) << 16);
}
__device__ __forceinline__ float gelu_f(float x) {
  return 0.5f * x * (1.0f + erff(x * 0.70710678118654752f));
}

// ---------- WMMA GEMM core ----------
// D[m][col] = sum_k A[m][k] * B[col][k]
// A: bf16 [M x K], row-major, leading dim lda (K innermost)
// B: bf16 [Ncols x K], row-major, leading dim ldb (K innermost) == column-major B operand
// Each wave computes MT x NT tiles of 16x16 starting at (mbase, nbase).
template <int MT, int NT>
__device__ __forceinline__ void gemm_acc(const us16* __restrict__ A,
                                         const us16* __restrict__ B,
                                         long lda, long ldb, int K,
                                         int mbase, int nbase,
                                         v8f (&acc)[MT][NT]) {
  const int lane = threadIdx.x & 31;
  const int ml   = lane & 15;
  const int hk   = lane >> 4;
#pragma unroll
  for (int t = 0; t < MT; ++t)
#pragma unroll
    for (int j = 0; j < NT; ++j)
      acc[t][j] = (v8f){0.f, 0.f, 0.f, 0.f, 0.f, 0.f, 0.f, 0.f};

  for (int kk = 0; kk < K; kk += 32) {
    Frag fb[NT];
#pragma unroll
    for (int j = 0; j < NT; ++j) {
      const us16* bp = B + (size_t)(nbase + 16 * j + ml) * (size_t)ldb + kk + 8 * hk;
      fb[j].q[0] = *(const u32x4*)(bp);
      fb[j].q[1] = *(const u32x4*)(bp + 16);
    }
#pragma unroll
    for (int t = 0; t < MT; ++t) {
      Frag fa;
      const us16* ap = A + (size_t)(mbase + 16 * t + ml) * (size_t)lda + kk + 8 * hk;
      fa.q[0] = *(const u32x4*)(ap);
      fa.q[1] = *(const u32x4*)(ap + 16);
#pragma unroll
      for (int j = 0; j < NT; ++j) {
        acc[t][j] = __builtin_amdgcn_wmma_f32_16x16x32_bf16(
            false, fa.v, false, fb[j].v, (short)0, acc[t][j], false, false);
      }
    }
  }
}

// ---------- problem constants ----------
#define BATCH 16
#define C256  256
#define MID   1536
#define NPB   4096            // spatial per batch (64*64)
#define NTOT  (BATCH * NPB)   // 65536
#define VPADR 384             // padded rows of v_pad (257 used, ones at 256)
#define VROWS 272             // 17 tiles of 16

// ---------- utility kernels ----------
__global__ void k_cast_bf16(const float* __restrict__ in, us16* __restrict__ out, int n) {
  int i = blockIdx.x * blockDim.x + threadIdx.x;
  if (i < n) out[i] = f2bf(in[i]);
}

// x[B,C,H,W] f32 -> xT[s][c] bf16 (NHWC), 32x32 LDS tile transpose
__global__ __launch_bounds__(256) void k_transpose_x(const float* __restrict__ x,
                                                     us16* __restrict__ xT) {
  __shared__ float t[32][33];
  const int b  = blockIdx.z;
  const int cb = blockIdx.y * 32;
  const int nb = blockIdx.x * 32;
  const int tx = threadIdx.x & 31;
  const int ty = threadIdx.x >> 5;   // 0..7
#pragma unroll
  for (int i = 0; i < 4; ++i) {
    int row = ty + 8 * i;            // channel within tile
    t[row][tx] = x[((size_t)b * C256 + cb + row) * NPB + nb + tx];
  }
  __syncthreads();
#pragma unroll
  for (int i = 0; i < 4; ++i) {
    int row = ty + 8 * i;            // spatial within tile
    xT[((size_t)b * NPB + nb + row) * C256 + cb + tx] = f2bf(t[tx][row]);
  }
}

// fill v_pad rows 256..383: row 256 = 1.0 (bf16 0x3F80), rows 257..383 = 0
__global__ void k_fill_vpad(us16* __restrict__ vp) {
  int i = blockIdx.x * blockDim.x + threadIdx.x;
  const int per_b = 128 * NPB;       // 2^19
  if (i >= BATCH * per_b) return;
  int b   = i >> 19;
  int rem = i & (per_b - 1);
  int row = 256 + (rem >> 12);
  int n   = rem & (NPB - 1);
  vp[((size_t)b * VPADR + row) * NPB + n] = (row == 256) ? (us16)0x3F80 : (us16)0;
}

// ---------- qkv GEMM: [768x256] @ xT -> q(NHWC,relu), k([C x Ntot],relu), v(v_pad [384 x 4096]/batch) ----------
__global__ __launch_bounds__(256) void k_gemm_qkv(const us16* __restrict__ W,
                                                  const us16* __restrict__ xT,
                                                  us16* __restrict__ qT,
                                                  us16* __restrict__ kC,
                                                  us16* __restrict__ vp) {
  const int wave = threadIdx.x >> 5;
  const int lane = threadIdx.x & 31;
  const int ml = lane & 15, hk = lane >> 4;
  const int mbase = blockIdx.x * 128 + (wave >> 2) * 64;
  const int nbase = blockIdx.y * 128 + (wave & 3) * 32;
  v8f acc[4][2];
  gemm_acc<4, 2>(W, xT, 256, 256, 256, mbase, nbase, acc);
#pragma unroll
  for (int ti = 0; ti < 4; ++ti) {
    const int oo = mbase + 16 * ti;        // tile base (16-aligned; regions at 0/256/512)
    const int o0 = oo + 8 * hk;            // first of 8 contiguous channels
#pragma unroll
    for (int tj = 0; tj < 2; ++tj) {
      const int s = nbase + 16 * tj + ml;
      if (oo < 256) {                      // q: relu, NHWC packed store
        u16x8 p;
#pragma unroll
        for (int r = 0; r < 8; ++r) p[r] = f2bf(fmaxf(acc[ti][tj][r], 0.f));
        *(u16x8*)(qT + (size_t)s * C256 + o0) = p;
      } else if (oo < 512) {               // k: relu, [C x Ntot]
#pragma unroll
        for (int r = 0; r < 8; ++r)
          kC[(size_t)(o0 + r - 256) * NTOT + s] = f2bf(fmaxf(acc[ti][tj][r], 0.f));
      } else {                             // v: per-batch v_pad [384 x 4096]
        const int b = s >> 12, n = s & (NPB - 1);
#pragma unroll
        for (int r = 0; r < 8; ++r)
          vp[((size_t)b * VPADR + (o0 + r - 512)) * NPB + n] = f2bf(acc[ti][tj][r]);
      }
    }
  }
}

// ---------- kv GEMM (per batch): kv[v][c] = sum_n vpad[v][n] * k[c][n] ----------
__global__ __launch_bounds__(256) void k_gemm_kv(const us16* __restrict__ vp,
                                                 const us16* __restrict__ kC,
                                                 float* __restrict__ kvf) {
  const int b = blockIdx.z;
  const int wave = threadIdx.x >> 5;
  const int lane = threadIdx.x & 31;
  const int ml = lane & 15, hk = lane >> 4;
  const int mbase = blockIdx.x * 128 + (wave >> 2) * 64;   // v rows (0..383)
  const int nbase = blockIdx.y * 128 + (wave & 3) * 32;    // c cols (0..255)
  const us16* A = vp + (size_t)b * VPADR * NPB;
  const us16* B = kC + (size_t)b * NPB;                    // ldb = NTOT
  v8f acc[4][2];
  gemm_acc<4, 2>(A, B, NPB, NTOT, NPB, mbase, nbase, acc);
#pragma unroll
  for (int ti = 0; ti < 4; ++ti)
#pragma unroll
    for (int tj = 0; tj < 2; ++tj) {
      const int col = nbase + 16 * tj + ml;
#pragma unroll
      for (int r = 0; r < 8; ++r) {
        const int m = mbase + 16 * ti + 8 * hk + r;
        if (m < VROWS)
          kvf[(size_t)b * VROWS * C256 + (size_t)m * C256 + col] = acc[ti][tj][r];
      }
    }
}

// ---------- attention out: out[v][n] = kv^T @ q, normalize by row 256, write attn NHWC bf16 ----------
__global__ __launch_bounds__(256) void k_attn(const us16* __restrict__ kvh,
                                              const us16* __restrict__ qT,
                                              us16* __restrict__ attn) {
  const int b = blockIdx.y;
  const int wave = threadIdx.x >> 5;
  const int lane = threadIdx.x & 31;
  const int ml = lane & 15, hk = lane >> 4;
  const int ncol = blockIdx.x * 128 + wave * 16;           // 16-col strip per wave
  const us16* A  = kvh + (size_t)b * VROWS * C256;         // [272 x 256]
  const us16* Bq = qT + (size_t)b * NPB * C256;            // ldb = 256

  v8f c[17];
#pragma unroll
  for (int t = 0; t < 17; ++t) c[t] = (v8f){0.f, 0.f, 0.f, 0.f, 0.f, 0.f, 0.f, 0.f};

  for (int kk = 0; kk < 256; kk += 32) {
    Frag fb;
    const us16* bp = Bq + (size_t)(ncol + ml) * C256 + kk + 8 * hk;
    fb.q[0] = *(const u32x4*)(bp);
    fb.q[1] = *(const u32x4*)(bp + 16);
#pragma unroll
    for (int t = 0; t < 17; ++t) {
      Frag fa;
      const us16* ap = A + (size_t)(16 * t + ml) * C256 + kk + 8 * hk;
      fa.q[0] = *(const u32x4*)(ap);
      fa.q[1] = *(const u32x4*)(ap + 16);
      c[t] = __builtin_amdgcn_wmma_f32_16x16x32_bf16(
          false, fa.v, false, fb.v, (short)0, c[t], false, false);
    }
  }
  // denominator: row 256 = tile 16, r=0, held by lanes 0..15; broadcast to both halves
  float dnum = c[16][0];
  float d    = __shfl(dnum, ml, 32) + 1e-5f;
  float rinv = 1.0f / d;
  const size_t srow = ((size_t)b * NPB + ncol + ml) * C256;
#pragma unroll
  for (int t = 0; t < 16; ++t) {
    u16x8 p;
#pragma unroll
    for (int r = 0; r < 8; ++r) p[r] = f2bf(c[t][r] * rinv);
    *(u16x8*)(attn + srow + 16 * t + 8 * hk) = p;
  }
}

// ---------- proj GEMM + BN + residual: y = x + BN(proj_w @ attn). y -> f32 NCHW + bf16 NHWC ----------
__global__ __launch_bounds__(256) void k_gemm_proj(const us16* __restrict__ W,
                                                   const us16* __restrict__ attn,
                                                   const float* __restrict__ x,
                                                   const float* __restrict__ g,
                                                   const float* __restrict__ bb,
                                                   const float* __restrict__ mm,
                                                   const float* __restrict__ vv,
                                                   float* __restrict__ yf,
                                                   us16* __restrict__ yT) {
  const int wave = threadIdx.x >> 5;
  const int lane = threadIdx.x & 31;
  const int ml = lane & 15, hk = lane >> 4;
  const int mbase = blockIdx.x * 128 + (wave >> 2) * 64;
  const int nbase = blockIdx.y * 128 + (wave & 3) * 32;
  v8f acc[4][2];
  gemm_acc<4, 2>(W, attn, 256, 256, 256, mbase, nbase, acc);
#pragma unroll
  for (int ti = 0; ti < 4; ++ti) {
    const int o0 = mbase + 16 * ti + 8 * hk;
#pragma unroll
    for (int tj = 0; tj < 2; ++tj) {
      const int s = nbase + 16 * tj + ml;
      const int b = s >> 12, n = s & (NPB - 1);
      u16x8 p;
#pragma unroll
      for (int r = 0; r < 8; ++r) {
        const int o = o0 + r;
        float sc = g[o] * rsqrtf(vv[o] + 1e-5f);
        float bi = bb[o] - mm[o] * sc;
        size_t xi = ((size_t)b * C256 + o) * NPB + n;
        float val = acc[ti][tj][r] * sc + bi + x[xi];
        yf[xi] = val;
        p[r] = f2bf(val);
      }
      *(u16x8*)(yT + (size_t)s * C256 + o0) = p;
    }
  }
}

// ---------- inv GEMM + bias + BN + GELU -> h1 NHWC bf16 ----------
__global__ __launch_bounds__(256) void k_gemm_inv(const us16* __restrict__ W,
                                                  const us16* __restrict__ yT,
                                                  const float* __restrict__ bias,
                                                  const float* __restrict__ g,
                                                  const float* __restrict__ beta,
                                                  const float* __restrict__ mm,
                                                  const float* __restrict__ vv,
                                                  us16* __restrict__ h1) {
  const int wave = threadIdx.x >> 5;
  const int lane = threadIdx.x & 31;
  const int ml = lane & 15, hk = lane >> 4;
  const int mbase = blockIdx.x * 128 + (wave >> 2) * 64;
  const int nbase = blockIdx.y * 128 + (wave & 3) * 32;
  v8f acc[4][2];
  gemm_acc<4, 2>(W, yT, 256, 256, 256, mbase, nbase, acc);
#pragma unroll
  for (int ti = 0; ti < 4; ++ti) {
    const int o0 = mbase + 16 * ti + 8 * hk;
#pragma unroll
    for (int tj = 0; tj < 2; ++tj) {
      const int s = nbase + 16 * tj + ml;
      u16x8 p;
#pragma unroll
      for (int r = 0; r < 8; ++r) {
        const int o = o0 + r;
        float sc = g[o] * rsqrtf(vv[o] + 1e-5f);
        float bi = beta[o] - mm[o] * sc;
        float val = gelu_f((acc[ti][tj][r] + bias[o]) * sc + bi);
        p[r] = f2bf(val);
      }
      *(u16x8*)(h1 + (size_t)s * MID + o0) = p;
    }
  }
}

// ---------- depthwise 3x3 (NHWC, 8 channels/thread) + bias + BN + GELU -> h2 ----------
__global__ void k_dwconv(const us16* __restrict__ h1, const float* __restrict__ w3,
                         const float* __restrict__ bias,
                         const float* __restrict__ g, const float* __restrict__ beta,
                         const float* __restrict__ mm, const float* __restrict__ vv,
                         us16* __restrict__ h2) {
  const int CV = MID / 8;                 // 192 channel-vectors
  long idx = (long)blockIdx.x * blockDim.x + threadIdx.x;
  if (idx >= (long)NTOT * CV) return;
  const int cvec = (int)(idx % CV);
  const int s    = (int)(idx / CV);
  const int c0 = cvec * 8;
  const int b  = s >> 12;
  const int hw = s & (NPB - 1);
  const int hh = hw >> 6, ww = hw & 63;
  float a[8];
#pragma unroll
  for (int j = 0; j < 8; ++j) a[j] = 0.f;
#pragma unroll
  for (int dh = -1; dh <= 1; ++dh) {
    const int h2i = hh + dh;
    if ((unsigned)h2i >= 64u) continue;
#pragma unroll
    for (int dw = -1; dw <= 1; ++dw) {
      const int w2i = ww + dw;
      if ((unsigned)w2i >= 64u) continue;
      const size_t s2 = ((size_t)b << 12) | ((size_t)h2i << 6) | (size_t)w2i;
      const u16x8 q = *(const u16x8*)(h1 + s2 * MID + c0);
      const int wi = (dh + 1) * 3 + (dw + 1);
#pragma unroll
      for (int j = 0; j < 8; ++j) a[j] += bf2f(q[j]) * w3[(size_t)(c0 + j) * 9 + wi];
    }
  }
  u16x8 p;
#pragma unroll
  for (int j = 0; j < 8; ++j) {
    const int c = c0 + j;
    float sc = g[c] * rsqrtf(vv[c] + 1e-5f);
    float bi = beta[c] - mm[c] * sc;
    p[j] = f2bf(gelu_f((a[j] + bias[c]) * sc + bi));
  }
  *(u16x8*)(h2 + (size_t)s * MID + c0) = p;
}

// ---------- pw GEMM + BN + GELU + residual -> d_out f32 NCHW ----------
__global__ __launch_bounds__(256) void k_gemm_pw(const us16* __restrict__ W,
                                                 const us16* __restrict__ h2,
                                                 const float* __restrict__ yf,
                                                 const float* __restrict__ g,
                                                 const float* __restrict__ beta,
                                                 const float* __restrict__ mm,
                                                 const float* __restrict__ vv,
                                                 float* __restrict__ out) {
  const int wave = threadIdx.x >> 5;
  const int lane = threadIdx.x & 31;
  const int ml = lane & 15, hk = lane >> 4;
  const int mbase = blockIdx.x * 128 + (wave >> 2) * 64;
  const int nbase = blockIdx.y * 128 + (wave & 3) * 32;
  v8f acc[4][2];
  gemm_acc<4, 2>(W, h2, MID, MID, MID, mbase, nbase, acc);
#pragma unroll
  for (int ti = 0; ti < 4; ++ti) {
    const int o0 = mbase + 16 * ti + 8 * hk;
#pragma unroll
    for (int tj = 0; tj < 2; ++tj) {
      const int s = nbase + 16 * tj + ml;
      const int b = s >> 12, n = s & (NPB - 1);
#pragma unroll
      for (int r = 0; r < 8; ++r) {
        const int o = o0 + r;
        float sc = g[o] * rsqrtf(vv[o] + 1e-5f);
        float bi = beta[o] - mm[o] * sc;
        size_t oi = ((size_t)b * C256 + o) * NPB + n;
        out[oi] = yf[oi] + gelu_f(acc[ti][tj][r] * sc + bi);
      }
    }
  }
}

// ---------- host side ----------
extern "C" void kernel_launch(void* const* d_in, const int* in_sizes, int n_in,
                              void* d_out, int out_size, void* d_ws, size_t ws_size,
                              hipStream_t stream) {
  (void)in_sizes; (void)n_in; (void)out_size; (void)ws_size;
  const float* x      = (const float*)d_in[0];
  const float* qkv_w  = (const float*)d_in[1];
  const float* proj_w = (const float*)d_in[2];
  const float* proj_g = (const float*)d_in[3];
  const float* proj_b = (const float*)d_in[4];
  const float* proj_m = (const float*)d_in[5];
  const float* proj_v = (const float*)d_in[6];
  const float* inv_w  = (const float*)d_in[7];
  const float* inv_b  = (const float*)d_in[8];
  const float* inv_g  = (const float*)d_in[9];
  const float* inv_be = (const float*)d_in[10];
  const float* inv_m  = (const float*)d_in[11];
  const float* inv_v  = (const float*)d_in[12];
  const float* dw_w   = (const float*)d_in[13];
  const float* dw_b   = (const float*)d_in[14];
  const float* dw_g   = (const float*)d_in[15];
  const float* dw_be  = (const float*)d_in[16];
  const float* dw_m   = (const float*)d_in[17];
  const float* dw_v   = (const float*)d_in[18];
  const float* pw_w   = (const float*)d_in[19];
  const float* pw_g   = (const float*)d_in[20];
  const float* pw_be  = (const float*)d_in[21];
  const float* pw_m   = (const float*)d_in[22];
  const float* pw_v   = (const float*)d_in[23];
  float* outp = (float*)d_out;

  // workspace layout (deterministic, recomputed each call)
  char* base = (char*)d_ws;
  size_t off = 0;
  auto take = [&](size_t bytes) -> char* {
    char* p = base + off;
    off = (off + bytes + 255) & ~(size_t)255;
    return p;
  };
  us16*  xT   = (us16*) take((size_t)NTOT * C256 * 2);
  us16*  qT   = (us16*) take((size_t)NTOT * C256 * 2);
  us16*  kC   = (us16*) take((size_t)NTOT * C256 * 2);
  us16*  vp   = (us16*) take((size_t)BATCH * VPADR * NPB * 2);
  float* kvf  = (float*)take((size_t)BATCH * VROWS * C256 * 4);
  us16*  kvh  = (us16*) take((size_t)BATCH * VROWS * C256 * 2);
  us16*  attn = (us16*) take((size_t)NTOT * C256 * 2);
  float* yf   = (float*)take((size_t)NTOT * C256 * 4);
  us16*  yT   = (us16*) take((size_t)NTOT * C256 * 2);
  us16*  h1   = (us16*) take((size_t)NTOT * MID * 2);
  us16*  h2   = (us16*) take((size_t)NTOT * MID * 2);
  us16*  wq   = (us16*) take((size_t)768 * 256 * 2);
  us16*  wp   = (us16*) take((size_t)256 * 256 * 2);
  us16*  wi   = (us16*) take((size_t)MID * 256 * 2);
  us16*  ww   = (us16*) take((size_t)256 * MID * 2);

  auto cgrid = [](int n) { return dim3((n + 255) / 256); };

  // weight casts (all already [O x C] row-major = WMMA A layout)
  k_cast_bf16<<<cgrid(768 * 256),  256, 0, stream>>>(qkv_w, wq, 768 * 256);
  k_cast_bf16<<<cgrid(256 * 256),  256, 0, stream>>>(proj_w, wp, 256 * 256);
  k_cast_bf16<<<cgrid(MID * 256),  256, 0, stream>>>(inv_w, wi, MID * 256);
  k_cast_bf16<<<cgrid(256 * MID),  256, 0, stream>>>(pw_w, ww, 256 * MID);

  // x -> NHWC bf16 ; v_pad filler rows
  k_transpose_x<<<dim3(NPB / 32, C256 / 32, BATCH), 256, 0, stream>>>(x, xT);
  k_fill_vpad<<<cgrid(BATCH * 128 * NPB), 256, 0, stream>>>(vp);

  // qkv projection
  k_gemm_qkv<<<dim3(768 / 128, NTOT / 128), 256, 0, stream>>>(wq, xT, qT, kC, vp);

  // kv = v_pad @ k^T (per batch)
  k_gemm_kv<<<dim3(VPADR / 128, C256 / 128, BATCH), 256, 0, stream>>>(vp, kC, kvf);
  k_cast_bf16<<<cgrid(BATCH * VROWS * C256), 256, 0, stream>>>(kvf, kvh, BATCH * VROWS * C256);

  // out = kv^T @ q, normalized
  k_attn<<<dim3(NPB / 128, BATCH), 256, 0, stream>>>(kvh, qT, attn);

  // proj + BN + residual
  k_gemm_proj<<<dim3(C256 / 128, NTOT / 128), 256, 0, stream>>>(
      wp, attn, x, proj_g, proj_b, proj_m, proj_v, yf, yT);

  // inverted bottleneck
  k_gemm_inv<<<dim3(MID / 128, NTOT / 128), 256, 0, stream>>>(
      wi, yT, inv_b, inv_g, inv_be, inv_m, inv_v, h1);

  long dwn = (long)NTOT * (MID / 8);
  k_dwconv<<<dim3((unsigned)((dwn + 255) / 256)), 256, 0, stream>>>(
      h1, dw_w, dw_b, dw_g, dw_be, dw_m, dw_v, h2);

  k_gemm_pw<<<dim3(C256 / 128, NTOT / 128), 256, 0, stream>>>(
      ww, h2, yf, pw_g, pw_be, pw_m, pw_v, outp);
}
